// BidirectionalAttentionModule_57140244906002
// MI455X (gfx1250) — compile-verified, matrix operand order checked by reading, not run on hardware
//
#include <hip/hip_runtime.h>
#include <stdint.h>

// ---------------------------------------------------------------------------
// BidirectionalAttentionModule for MI455X (gfx1250, wave32, WMMA).
// Heavy GEMMs (146 GFLOP) via v_wmma_f32_16x16x32_bf16; operand staging via
// global_load_async_to_lds_b128 (ASYNCcnt) with double-buffered LDS.
// ---------------------------------------------------------------------------

typedef __attribute__((ext_vector_type(16))) __bf16 v16bf;
typedef __attribute__((ext_vector_type(8)))  __bf16 v8bf;
typedef __attribute__((ext_vector_type(8)))  float  v8f;

#define BATCH   16
#define LCOMP   512
#define LPROT   2048
#define DDIM    256
#define NLAYERS 4

#define TILE    128
#define KSTEP   32
#define LDS_LD  40   // 32 bf16 + 8 pad -> 80B row stride, conflict-free b128 LDS reads

__device__ __forceinline__ unsigned short f32_to_bf16(float f) {
  union { float f; uint32_t u; } v; v.f = f;
  uint32_t u = v.u;
  return (unsigned short)((u + 0x7FFFu + ((u >> 16) & 1u)) >> 16);  // RNE
}

// Async 16-byte global -> LDS copy (per-lane), tracked by ASYNCcnt.
__device__ __forceinline__ void async_cp16(unsigned short* lds,
                                           const unsigned short* g) {
  asm volatile("global_load_async_to_lds_b128 %0, %1, off"
               :: "v"((unsigned)(unsigned long long)lds),
                  "v"((unsigned long long)g)
               : "memory");
}

__device__ __forceinline__ void wait_async0() {
  asm volatile("s_wait_asynccnt 0x0" ::: "memory");
}

// Fragment load per ISA 16-bit A-matrix 16x32 layout:
// lane l: row = base + (l&15); elems 0..7 = K(8*khalf .. 8*khalf+7),
// elems 8..15 = K(16+8*khalf .. 23+8*khalf), khalf = l>>4.
__device__ __forceinline__ v16bf load_frag(const unsigned short* rowp, int khalf) {
  const v8bf lo = *reinterpret_cast<const v8bf*>(rowp + khalf * 8);
  const v8bf hi = *reinterpret_cast<const v8bf*>(rowp + khalf * 8 + 16);
  v16bf r;
#pragma unroll
  for (int e = 0; e < 8; ++e) { r[e] = lo[e]; r[e + 8] = hi[e]; }
  return r;
}

// out(M x N) = X(M x K) @ Y(N x K)^T  [+bias] [tanh] [mask] -> f32 / bf16 / bf16-transposed
// BIAS_MODE: 0 none, 1 bias[m], 2 bias[n].  MASKED: S-kernel epilogue
// (tanh * cm[m]*pm[n], writes A and A^T in bf16).
template <int BIAS_MODE, bool TANH, bool OUT_F32, bool OUT_BF16, bool MASKED>
__global__ __launch_bounds__(256) void gemm_wmma(
    const unsigned short* __restrict__ X, long long xBS, int ldx,
    const unsigned short* __restrict__ Y, long long yBS, int ldy,
    const float* __restrict__ bias,
    float* __restrict__ outF, long long oFBS, int ldoF,
    unsigned short* __restrict__ outB, long long oBBS, int ldoB,
    unsigned short* __restrict__ outT, long long oTBS, int ldoT,
    const int* __restrict__ cmask, const int* __restrict__ pmask,
    int K) {
  __shared__ unsigned short Xs[2][TILE * LDS_LD];
  __shared__ unsigned short Ys[2][TILE * LDS_LD];

  const int b    = blockIdx.z;
  const int r0   = blockIdx.y * TILE;
  const int c0   = blockIdx.x * TILE;
  const int tid  = threadIdx.x;
  const int lane = tid & 31;
  const int wave = tid >> 5;
  const int wm   = wave & 3;   // 4 waves along M (4*32 = 128)
  const int wn   = wave >> 2;  // 2 waves along N (2*64 = 128)
  const int lrow = lane & 15;
  const int khalf = lane >> 4;

  const unsigned short* Xb = X + (long long)b * xBS;
  const unsigned short* Yb = Y + (long long)b * yBS;

  // Per-thread staging coordinates: 2 X chunks + 2 Y chunks of 16B per K-step.
  const int srow0 = tid >> 2;        // 0..63
  const int srow1 = srow0 + 64;      // 64..127
  const int skc   = (tid & 3) * 8;   // element offset within the 32-wide K slab

  auto stage = [&](int kk, int buf) {
    async_cp16(&Xs[buf][srow0 * LDS_LD + skc],
               Xb + (long long)(r0 + srow0) * ldx + kk + skc);
    async_cp16(&Xs[buf][srow1 * LDS_LD + skc],
               Xb + (long long)(r0 + srow1) * ldx + kk + skc);
    async_cp16(&Ys[buf][srow0 * LDS_LD + skc],
               Yb + (long long)(c0 + srow0) * ldy + kk + skc);
    async_cp16(&Ys[buf][srow1 * LDS_LD + skc],
               Yb + (long long)(c0 + srow1) * ldy + kk + skc);
  };

  v8f acc[2][4];
#pragma unroll
  for (int i = 0; i < 2; ++i)
#pragma unroll
    for (int j = 0; j < 4; ++j)
#pragma unroll
      for (int e = 0; e < 8; ++e) acc[i][j][e] = 0.0f;

  // Prologue: stage tile 0 into buffer 0.
  stage(0, 0);
  wait_async0();
  __syncthreads();

  const int nstep = K / KSTEP;
  for (int s = 0; s < nstep; ++s) {
    const int cur = s & 1;
    // Stage next tile into the other buffer while we compute on this one.
    if (s + 1 < nstep) stage((s + 1) * KSTEP, cur ^ 1);

    v16bf af[2], bfr[4];
#pragma unroll
    for (int ti = 0; ti < 2; ++ti)
      af[ti] = load_frag(&Xs[cur][(wm * 32 + ti * 16 + lrow) * LDS_LD], khalf);
#pragma unroll
    for (int tj = 0; tj < 4; ++tj)
      bfr[tj] = load_frag(&Ys[cur][(wn * 64 + tj * 16 + lrow) * LDS_LD], khalf);

#pragma unroll
    for (int ti = 0; ti < 2; ++ti)
#pragma unroll
      for (int tj = 0; tj < 4; ++tj)
        acc[ti][tj] = __builtin_amdgcn_wmma_f32_16x16x32_bf16(
            false, af[ti], false, bfr[tj], (short)0, acc[ti][tj], false, false);

    if (s + 1 < nstep) {
      wait_async0();     // our async writes for tile s+1 have landed in LDS
      __syncthreads();   // all waves' writes visible; prior ds_loads complete
    }
  }

  // Epilogue.  C/D layout: VGPR v, lane l -> m = v + 8*(l>>4), n = l&15.
  const int nlane = lane & 15;
  const int moff  = (lane >> 4) * 8;
#pragma unroll
  for (int ti = 0; ti < 2; ++ti) {
#pragma unroll
    for (int tj = 0; tj < 4; ++tj) {
      const int mb = r0 + wm * 32 + ti * 16 + moff;
      const int n  = c0 + wn * 64 + tj * 16 + nlane;
#pragma unroll
      for (int v = 0; v < 8; ++v) {
        const int m = mb + v;
        float x = acc[ti][tj][v];
        if (BIAS_MODE == 1) x += bias[m];
        if (BIAS_MODE == 2) x += bias[n];
        if (TANH) x = tanhf(x);
        if (MASKED) {
          const float msk =
              (float)(cmask[b * LCOMP + m] * pmask[b * LPROT + n]);
          x = tanhf(x) * msk;
          const unsigned short hv = f32_to_bf16(x);
          outB[(long long)b * oBBS + (long long)m * ldoB + n] = hv;
          outT[(long long)b * oTBS + (long long)n * ldoT + m] = hv;
        } else {
          if (OUT_F32)
            outF[(long long)b * oFBS + (long long)m * ldoF + n] = x;
          if (OUT_BF16)
            outB[(long long)b * oBBS + (long long)m * ldoB + n] = f32_to_bf16(x);
        }
      }
    }
  }
}

// --------------------------- small helper kernels ---------------------------

__global__ __launch_bounds__(256) void cvt_bf16(const float* __restrict__ src,
                                                unsigned short* __restrict__ dst,
                                                long long n) {
  long long i = (long long)blockIdx.x * 256 + threadIdx.x;
  if (i < n) dst[i] = f32_to_bf16(src[i]);
}

// Wt[l][n][k] = bf16(W[l][k][n]);  W is [L, D, D]
__global__ __launch_bounds__(256) void transpose_w_bf16(
    const float* __restrict__ W, unsigned short* __restrict__ Wt) {
  const int i = blockIdx.x * 256 + threadIdx.x;  // < L*D*D = 262144
  const int l = i >> 16;
  const int rem = i & 65535;
  const int n = rem >> 8, k = rem & 255;
  Wt[i] = f32_to_bf16(W[(l << 16) + (k << 8) + n]);
}

// Per (batch, layer): scores over the layerwise slice, masked softmax,
// then pooled feature (D=256, one output per thread).
__global__ __launch_bounds__(256) void attn_pool(
    const float* __restrict__ layerwise,  // [B, N, 2*D*L]
    const float* __restrict__ feat,       // [B, N, D]
    const int* __restrict__ mask,         // [B, N]
    const float* __restrict__ Wa,         // [L, 2D]
    const float* __restrict__ ba,         // [L]
    float* __restrict__ pools,            // [B, L*D]
    int N) {
  __shared__ float sc[LPROT];
  __shared__ float red[256];
  const int b = blockIdx.x, layer = blockIdx.y, tid = threadIdx.x;
  const float* lw = layerwise + (long long)b * N * (2 * DDIM * NLAYERS) +
                    layer * (2 * DDIM);
  const float* wv  = Wa + layer * (2 * DDIM);
  const float  bsc = ba[layer];

  for (int c = tid; c < N; c += 256) {
    const float* row = lw + (long long)c * (2 * DDIM * NLAYERS);
    float s = 0.f;
    for (int d2 = 0; d2 < 2 * DDIM; ++d2) s = fmaf(row[d2], wv[d2], s);
    sc[c] = s + bsc;
  }
  __syncthreads();

  float mx = -3.4e38f;
  for (int c = tid; c < N; c += 256) mx = fmaxf(mx, sc[c]);
  red[tid] = mx;
  __syncthreads();
  for (int s = 128; s > 0; s >>= 1) {
    if (tid < s) red[tid] = fmaxf(red[tid], red[tid + s]);
    __syncthreads();
  }
  const float rowmax = red[0];
  __syncthreads();

  float sm = 0.f;
  for (int c = tid; c < N; c += 256) {
    const float e = __expf(sc[c] - rowmax) * (float)mask[b * N + c];
    sc[c] = e;
    sm += e;
  }
  red[tid] = sm;
  __syncthreads();
  for (int s = 128; s > 0; s >>= 1) {
    if (tid < s) red[tid] += red[tid + s];
    __syncthreads();
  }
  const float denom = red[0] + 1e-6f;
  __syncthreads();

  const float* fb = feat + (long long)b * N * DDIM;
  float acc = 0.f;
  for (int c = 0; c < N; ++c)
    acc = fmaf(fb[(long long)c * DDIM + tid], sc[c], acc);
  pools[b * (NLAYERS * DDIM) + layer * DDIM + tid] = acc / denom;
}

// out[b, d] = pools[b, :] @ W[:, d] + bias[d];  K = L*D = 1024
__global__ __launch_bounds__(256) void final_linear(
    const float* __restrict__ pools, const float* __restrict__ W,
    const float* __restrict__ bias, float* __restrict__ out) {
  const int b = blockIdx.x, d2 = threadIdx.x;
  float acc = bias[d2];
  const float* p = pools + b * (NLAYERS * DDIM);
  for (int k = 0; k < NLAYERS * DDIM; ++k)
    acc = fmaf(p[k], W[k * DDIM + d2], acc);
  out[b * DDIM + d2] = acc;
}

// ------------------------------- launch --------------------------------------

extern "C" void kernel_launch(void* const* d_in, const int* in_sizes, int n_in,
                              void* d_out, int out_size, void* d_ws,
                              size_t ws_size, hipStream_t stream) {
  const float* comp_feat = (const float*)d_in[0];
  const int*   comp_mask = (const int*)d_in[1];
  const float* prot_feat = (const float*)d_in[2];
  const int*   prot_mask = (const int*)d_in[3];
  const float* U     = (const float*)d_in[4];
  const float* W_p2c = (const float*)d_in[5];
  const float* b_p2c = (const float*)d_in[6];
  const float* W_c2p = (const float*)d_in[7];
  const float* b_c2p = (const float*)d_in[8];
  const float* W_hc  = (const float*)d_in[9];
  const float* b_hc  = (const float*)d_in[10];
  const float* W_hp  = (const float*)d_in[11];
  const float* b_hp  = (const float*)d_in[12];
  const float* W_ac  = (const float*)d_in[13];
  const float* b_ac  = (const float*)d_in[14];
  const float* W_ap  = (const float*)d_in[15];
  const float* b_ap  = (const float*)d_in[16];
  const float* W_cc  = (const float*)d_in[17];
  const float* b_cc  = (const float*)d_in[18];
  const float* W_cp  = (const float*)d_in[19];
  const float* b_cp  = (const float*)d_in[20];

  float* comp_final = (float*)d_out;                       // [16, 256]
  float* prot_final = comp_final + BATCH * DDIM;           // [16, 256]
  float* clw = prot_final + BATCH * DDIM;                  // [16, 512, 2048]
  float* plw = clw + (size_t)BATCH * LCOMP * 2048;         // [16, 2048, 2048]

  // Workspace layout
  char* ws = (char*)d_ws;
  size_t off = 0;
  auto alloc = [&](size_t bytes) -> void* {
    void* p = ws + off;
    off += (bytes + 255) & ~(size_t)255;
    return p;
  };
  unsigned short* Cbf   = (unsigned short*)alloc((size_t)BATCH * LCOMP * DDIM * 2);
  unsigned short* Pbf   = (unsigned short*)alloc((size_t)BATCH * LPROT * DDIM * 2);
  unsigned short* Ut    = (unsigned short*)alloc((size_t)NLAYERS * DDIM * DDIM * 2);
  unsigned short* Wp2cT = (unsigned short*)alloc((size_t)NLAYERS * DDIM * DDIM * 2);
  unsigned short* Wc2pT = (unsigned short*)alloc((size_t)NLAYERS * DDIM * DDIM * 2);
  unsigned short* WhcT  = (unsigned short*)alloc((size_t)NLAYERS * DDIM * DDIM * 2);
  unsigned short* WhpT  = (unsigned short*)alloc((size_t)NLAYERS * DDIM * DDIM * 2);
  unsigned short* CUbf  = (unsigned short*)alloc((size_t)BATCH * LCOMP * DDIM * 2);
  unsigned short* PtilT = (unsigned short*)alloc((size_t)BATCH * DDIM * LPROT * 2);
  unsigned short* CtilT = (unsigned short*)alloc((size_t)BATCH * DDIM * LCOMP * 2);
  unsigned short* Abf   = (unsigned short*)alloc((size_t)BATCH * LCOMP * LPROT * 2);
  unsigned short* ATbf  = (unsigned short*)alloc((size_t)BATCH * LPROT * LCOMP * 2);
  float* cpool = (float*)alloc((size_t)BATCH * NLAYERS * DDIM * 4);
  float* ppool = (float*)alloc((size_t)BATCH * NLAYERS * DDIM * 4);

  const dim3 blk(256);

  // Convert features + transpose/convert weights to bf16.
  cvt_bf16<<<(BATCH * LCOMP * DDIM) / 256, blk, 0, stream>>>(
      comp_feat, Cbf, (long long)BATCH * LCOMP * DDIM);
  cvt_bf16<<<(BATCH * LPROT * DDIM) / 256, blk, 0, stream>>>(
      prot_feat, Pbf, (long long)BATCH * LPROT * DDIM);
  transpose_w_bf16<<<(NLAYERS * DDIM * DDIM) / 256, blk, 0, stream>>>(U, Ut);
  transpose_w_bf16<<<(NLAYERS * DDIM * DDIM) / 256, blk, 0, stream>>>(W_p2c, Wp2cT);
  transpose_w_bf16<<<(NLAYERS * DDIM * DDIM) / 256, blk, 0, stream>>>(W_c2p, Wc2pT);
  transpose_w_bf16<<<(NLAYERS * DDIM * DDIM) / 256, blk, 0, stream>>>(W_hc, WhcT);
  transpose_w_bf16<<<(NLAYERS * DDIM * DDIM) / 256, blk, 0, stream>>>(W_hp, WhpT);

  for (int i = 0; i < NLAYERS; ++i) {
    const size_t wOff = (size_t)i * DDIM * DDIM;

    // CU[b] = C @ U[i]          (512x256 = [512x256]@[256x256]^T, bf16 out)
    gemm_wmma<0, false, false, true, false><<<dim3(2, 4, BATCH), blk, 0, stream>>>(
        Cbf, (long long)LCOMP * DDIM, DDIM, Ut + wOff, 0, DDIM, nullptr,
        nullptr, 0, 0, CUbf, (long long)LCOMP * DDIM, DDIM,
        nullptr, 0, 0, nullptr, nullptr, DDIM);

    // PtilT[b] = tanh(W_p2c^T @ P^T + b)   (256x2048, bf16, bias per-row)
    gemm_wmma<1, true, false, true, false><<<dim3(16, 2, BATCH), blk, 0, stream>>>(
        Wp2cT + wOff, 0, DDIM, Pbf, (long long)LPROT * DDIM, DDIM,
        b_p2c + i * DDIM, nullptr, 0, 0,
        PtilT, (long long)DDIM * LPROT, LPROT,
        nullptr, 0, 0, nullptr, nullptr, DDIM);

    // CtilT[b] = tanh(W_c2p^T @ C^T + b)   (256x512, bf16, bias per-row)
    gemm_wmma<1, true, false, true, false><<<dim3(4, 2, BATCH), blk, 0, stream>>>(
        Wc2pT + wOff, 0, DDIM, Cbf, (long long)LCOMP * DDIM, DDIM,
        b_c2p + i * DDIM, nullptr, 0, 0,
        CtilT, (long long)DDIM * LCOMP, LCOMP,
        nullptr, 0, 0, nullptr, nullptr, DDIM);

    // Hc -> comp_layerwise[:, :, i*512 .. i*512+256)   (f32, tanh, bias per-col)
    gemm_wmma<2, true, true, false, false><<<dim3(2, 4, BATCH), blk, 0, stream>>>(
        Cbf, (long long)LCOMP * DDIM, DDIM, WhcT + wOff, 0, DDIM,
        b_hc + i * DDIM,
        clw + (size_t)i * 512, (long long)LCOMP * 2048, 2048,
        nullptr, 0, 0, nullptr, 0, 0, nullptr, nullptr, DDIM);

    // Hp -> prot_layerwise[:, :, i*512 .. i*512+256)
    gemm_wmma<2, true, true, false, false><<<dim3(2, 16, BATCH), blk, 0, stream>>>(
        Pbf, (long long)LPROT * DDIM, DDIM, WhpT + wOff, 0, DDIM,
        b_hp + i * DDIM,
        plw + (size_t)i * 512, (long long)LPROT * 2048, 2048,
        nullptr, 0, 0, nullptr, 0, 0, nullptr, nullptr, DDIM);

    // A = tanh(CU @ P^T) * mask   -> Abf [c,p] and ATbf [p,c] (bf16)
    gemm_wmma<0, false, false, false, true><<<dim3(16, 4, BATCH), blk, 0, stream>>>(
        CUbf, (long long)LCOMP * DDIM, DDIM, Pbf, (long long)LPROT * DDIM, DDIM,
        nullptr, nullptr, 0, 0,
        Abf, (long long)LCOMP * LPROT, LPROT,
        ATbf, (long long)LPROT * LCOMP, LCOMP,
        comp_mask, prot_mask, DDIM);

    // comp_cross = A @ Ptil  -> comp_layerwise[:, :, i*512+256 ..)
    gemm_wmma<0, false, true, false, false><<<dim3(2, 4, BATCH), blk, 0, stream>>>(
        Abf, (long long)LCOMP * LPROT, LPROT,
        PtilT, (long long)DDIM * LPROT, LPROT, nullptr,
        clw + (size_t)i * 512 + 256, (long long)LCOMP * 2048, 2048,
        nullptr, 0, 0, nullptr, 0, 0, nullptr, nullptr, LPROT);

    // prot_cross = A^T @ Ctil -> prot_layerwise[:, :, i*512+256 ..)
    gemm_wmma<0, false, true, false, false><<<dim3(2, 16, BATCH), blk, 0, stream>>>(
        ATbf, (long long)LPROT * LCOMP, LCOMP,
        CtilT, (long long)DDIM * LCOMP, LCOMP, nullptr,
        plw + (size_t)i * 512 + 256, (long long)LPROT * 2048, 2048,
        nullptr, 0, 0, nullptr, 0, 0, nullptr, nullptr, LCOMP);
  }

  // Masked-softmax attention pooling per (batch, layer), then final linears.
  attn_pool<<<dim3(BATCH, NLAYERS), blk, 0, stream>>>(
      clw, comp_feat, comp_mask, W_ac, b_ac, cpool, LCOMP);
  attn_pool<<<dim3(BATCH, NLAYERS), blk, 0, stream>>>(
      plw, prot_feat, prot_mask, W_ap, b_ap, ppool, LPROT);
  final_linear<<<BATCH, blk, 0, stream>>>(cpool, W_cc, b_cc, comp_final);
  final_linear<<<BATCH, blk, 0, stream>>>(ppool, W_cp, b_cp, prot_final);
}
